// QuantizedRNN_73684458930246
// MI455X (gfx1250) — matrix operand order repaired and checked by reading, not compiled, not run on hardware
//
#include <hip/hip_runtime.h>
#include <hip/hip_bf16.h>
#include <stdint.h>

// Problem constants (from the reference setup_inputs)
#define RNN_B 128
#define RNN_S 512
#define RNN_I 256
#define RNN_H 512

typedef __bf16 bf16_t;
typedef bf16_t v16bf __attribute__((ext_vector_type(16)));
typedef bf16_t v8bf  __attribute__((ext_vector_type(8)));
typedef float  v8f   __attribute__((ext_vector_type(8)));
typedef float  v2f   __attribute__((ext_vector_type(2)));

union V16U { v16bf v; v8bf h[2]; };

// ---------------------------------------------------------------------------
// Phase 1: init — zero the device barrier counter, seed h ping buffer with h0
// ---------------------------------------------------------------------------
__global__ void init_kernel(const float* __restrict__ h0,
                            float* __restrict__ h_ping,
                            unsigned* __restrict__ bar)
{
    const int i = blockIdx.x * blockDim.x + threadIdx.x;
    if (i == 0) *bar = 0u;
    if (i < RNN_B * RNN_H) h_ping[i] = h0[i];
}

// ---------------------------------------------------------------------------
// Phase 2: pre[m, n] = qx[m, :] . qw_ih[n, :] + qb_ih[n] + qb_hh[n]
//   M = B*S = 65536, N = H = 512, K = I = 256
//   bf16 WMMA (fquant E8M7 == bf16 RNE); fp32 accumulate; writes into d_out.
// ---------------------------------------------------------------------------
#define BM 128
#define BN 128
#define BK 32
#define ASTR (BK + 8)   // LDS row stride in bf16 elems (80B: keeps 16B align, skews banks)

__global__ __launch_bounds__(256)
void gemm_pre_kernel(const float* __restrict__ x,
                     const float* __restrict__ w,      // w_ih [H, I]
                     const float* __restrict__ b_ih,
                     const float* __restrict__ b_hh,
                     float* __restrict__ out)          // [M, H] region of d_out
{
    __shared__ alignas(16) bf16_t As[BM][ASTR];
    __shared__ alignas(16) bf16_t Bs[BN][ASTR];

    const int tid  = threadIdx.x;
    const int lane = tid & 31;
    const int wv   = tid >> 5;       // 8 waves (wave32)
    const int wm   = wv & 3;         // 4 waves along M -> 32 rows each
    const int wn   = wv >> 2;        // 2 waves along N -> 64 cols each

    const int mblk = blockIdx.x * BM;
    const int nblk = blockIdx.y * BN;

    v8f acc[2][4];
    #pragma unroll
    for (int i = 0; i < 2; ++i)
        #pragma unroll
        for (int j = 0; j < 4; ++j)
            acc[i][j] = v8f{};

    const int lrow  = tid >> 1;          // 0..127
    const int lhalf = (tid & 1) << 4;    // 0 or 16

    for (int kc = 0; kc < RNN_I; kc += BK) {
        // Stage tiles, quantizing fp32 -> bf16 (RNE) on the fly
        const float* ap = x + (size_t)(mblk + lrow) * RNN_I + kc + lhalf;
        const float* bp = w + (size_t)(nblk + lrow) * RNN_I + kc + lhalf;
        #pragma unroll
        for (int j = 0; j < 16; j += 4) {
            float4 fa = *(const float4*)(ap + j);
            float4 fb = *(const float4*)(bp + j);
            As[lrow][lhalf + j + 0] = (bf16_t)fa.x;
            As[lrow][lhalf + j + 1] = (bf16_t)fa.y;
            As[lrow][lhalf + j + 2] = (bf16_t)fa.z;
            As[lrow][lhalf + j + 3] = (bf16_t)fa.w;
            Bs[lrow][lhalf + j + 0] = (bf16_t)fb.x;
            Bs[lrow][lhalf + j + 1] = (bf16_t)fb.y;
            Bs[lrow][lhalf + j + 2] = (bf16_t)fb.z;
            Bs[lrow][lhalf + j + 3] = (bf16_t)fb.w;
        }
        __syncthreads();

        // CDNA5 16-bit fragment layouts (ISA 7.12.2)
        const int koff  = (lane >> 4) << 3;   // A: 0 / 8
        const int kbase = (lane >> 4) << 4;   // unused; B reads at koff (0 / 8) like A
        V16U afrag[2], bfrag[4];
        #pragma unroll
        for (int mi = 0; mi < 2; ++mi) {
            const bf16_t* rp = &As[wm * 32 + mi * 16 + (lane & 15)][0];
            afrag[mi].h[0] = *(const v8bf*)(rp + koff);        // K = koff..koff+7
            afrag[mi].h[1] = *(const v8bf*)(rp + 16 + koff);   // K = 16+koff..
        }
        #pragma unroll
        for (int ni = 0; ni < 4; ++ni) {
            const bf16_t* rp = &Bs[wn * 64 + ni * 16 + (lane & 15)][0];
            bfrag[ni].h[0] = *(const v8bf*)(rp + koff);        // K = koff..koff+7
            bfrag[ni].h[1] = *(const v8bf*)(rp + 16 + koff);
        }
        #pragma unroll
        for (int mi = 0; mi < 2; ++mi)
            #pragma unroll
            for (int ni = 0; ni < 4; ++ni)
                acc[mi][ni] = __builtin_amdgcn_wmma_f32_16x16x32_bf16(
                    false, afrag[mi].v, false, bfrag[ni].v,
                    (short)0, acc[mi][ni], false, false);
        __syncthreads();
    }

    // Epilogue: per-lane column n (C/D layout), add quantized biases, store f32
    #pragma unroll
    for (int ni = 0; ni < 4; ++ni) {
        const int n = nblk + wn * 64 + ni * 16 + (lane & 15);
        const float bias = (float)(bf16_t)b_ih[n] + (float)(bf16_t)b_hh[n];
        #pragma unroll
        for (int mi = 0; mi < 2; ++mi) {
            const int mrow = mblk + wm * 32 + mi * 16 + ((lane >> 4) << 3);
            #pragma unroll
            for (int j = 0; j < 8; ++j)
                out[(size_t)(mrow + j) * RNN_H + n] = acc[mi][ni][j] + bias;
        }
    }
}

// ---------------------------------------------------------------------------
// Phase 3: persistent recurrence — 16 workgroups (2 along B, 8 along H),
//   W_hh slice (dequantized bf16 values, f32) resident in LDS for all 512
//   steps; h kept fp32 (reference does NOT quantize h) -> f32 WMMA 16x16x4;
//   device-wide accumulating atomic barrier between steps; h ping-pongs in ws.
// ---------------------------------------------------------------------------
#define WSTR (RNN_H + 4)   // 516 f32: LDS row stride for W slice
#define HSTR 68            // f32 row stride for h chunk
#define NWG  16

__global__ __launch_bounds__(256)
void rnn_steps_kernel(const float* __restrict__ w_hh,
                      float* __restrict__ out,       // d_out: [B,S,H] + [B,H] tail
                      float* __restrict__ h_ping,
                      float* __restrict__ h_pong,
                      unsigned* __restrict__ bar)
{
    extern __shared__ float smem[];
    float* Wt = smem;                 // [64][WSTR]  (128+ KB)
    float* Ht = smem + 64 * WSTR;     // [64][HSTR]  (~17 KB)

    const int tid  = threadIdx.x;
    const int lane = tid & 31;
    const int wv   = tid >> 5;
    const int wvM  = wv & 1;          // 2 waves along M -> 32 rows
    const int wvN  = wv >> 1;         // 4 waves along N -> 16 cols

    const int wgM = blockIdx.x & 1;   // 2 workgroups along B (64 rows each)
    const int wgN = blockIdx.x >> 1;  // 8 workgroups along H (64 cols each)

    // Load + quantize the persistent W_hh slice once: rows [wgN*64, +64), K=512
    for (int idx = tid; idx < 64 * RNN_H; idx += 256) {
        const int r = idx >> 9;
        const int k = idx & (RNN_H - 1);
        Wt[r * WSTR + k] =
            (float)(bf16_t)w_hh[(size_t)(wgN * 64 + r) * RNN_H + k];
    }
    __syncthreads();

    float* cur = h_ping;
    float* nxt = h_pong;

    for (int t = 0; t < RNN_S; ++t) {
        v8f acc0 = v8f{}, acc1 = v8f{};

        for (int kc = 0; kc < RNN_H; kc += 64) {
            // Stage h chunk [64 rows x 64 k] (fp32, unquantized)
            {
                const int r   = tid >> 2;
                const int seg = (tid & 3) << 4;
                const float* hp =
                    cur + (size_t)(wgM * 64 + r) * RNN_H + kc + seg;
                #pragma unroll
                for (int j = 0; j < 16; j += 4) {
                    float4 f = *(const float4*)(hp + j);
                    Ht[r * HSTR + seg + j + 0] = f.x;
                    Ht[r * HSTR + seg + j + 1] = f.y;
                    Ht[r * HSTR + seg + j + 2] = f.z;
                    Ht[r * HSTR + seg + j + 3] = f.w;
                }
            }
            __syncthreads();

            const int m0  = wvM * 32 + (lane & 15);
            const int nn  = wvN * 16 + (lane & 15);
            const int klo = (lane >> 4) << 1;   // f32 A/B: K = klo, klo+1
            #pragma unroll
            for (int k4 = 0; k4 < 64; k4 += 4) {
                v2f a0 = *(const v2f*)&Ht[m0 * HSTR + k4 + klo];
                v2f a1 = *(const v2f*)&Ht[(m0 + 16) * HSTR + k4 + klo];
                v2f bb = *(const v2f*)&Wt[nn * WSTR + kc + k4 + klo];
                acc0 = __builtin_amdgcn_wmma_f32_16x16x4_f32(
                    false, a0, false, bb, (short)0, acc0, false, false);
                acc1 = __builtin_amdgcn_wmma_f32_16x16x4_f32(
                    false, a1, false, bb, (short)0, acc1, false, false);
            }
            __syncthreads();
        }

        // Epilogue: lin = pre (in d_out) + h@W^T, tanh, write out + next h
        const int n = wgN * 64 + wvN * 16 + (lane & 15);
        #pragma unroll
        for (int mi = 0; mi < 2; ++mi) {
            const v8f a = mi ? acc1 : acc0;
            const int mbase = wgM * 64 + wvM * 32 + mi * 16 + ((lane >> 4) << 3);
            #pragma unroll
            for (int j = 0; j < 8; ++j) {
                const int b = mbase + j;               // batch row 0..127
                const size_t oidx = ((size_t)b * RNN_S + t) * RNN_H + n;
                const float v = tanhf(a[j] + out[oidx]);
                out[oidx] = v;
                nxt[(size_t)b * RNN_H + n] = v;
                if (t == RNN_S - 1)
                    out[(size_t)RNN_B * RNN_S * RNN_H +
                        (size_t)b * RNN_H + n] = v;    // h_final tail
            }
        }

        // Device-wide step barrier (16 co-resident WGs; accumulating target,
        // counter zeroed by init_kernel each launch -> graph-replay safe)
        __threadfence();
        __syncthreads();
        if (tid == 0) {
            atomicAdd(bar, 1u);
            const unsigned target = (unsigned)NWG * (unsigned)(t + 1);
            while (atomicAdd(bar, 0u) < target)
                __builtin_amdgcn_s_sleep(2);
        }
        __syncthreads();
        __threadfence();

        float* tmp = cur; cur = nxt; nxt = tmp;
    }
}

// ---------------------------------------------------------------------------
extern "C" void kernel_launch(void* const* d_in, const int* in_sizes, int n_in,
                              void* d_out, int out_size, void* d_ws, size_t ws_size,
                              hipStream_t stream)
{
    const float* x    = (const float*)d_in[0];   // [B, S, I]
    const float* h0   = (const float*)d_in[1];   // [1, B, H]
    const float* w_ih = (const float*)d_in[2];   // [H, I]
    const float* w_hh = (const float*)d_in[3];   // [H, H]
    const float* b_ih = (const float*)d_in[4];   // [H]
    const float* b_hh = (const float*)d_in[5];   // [H]
    float* out = (float*)d_out;

    // Workspace layout: h ping (256 KB) | h pong (256 KB) | barrier counter
    float* h_ping = (float*)d_ws;
    float* h_pong = h_ping + RNN_B * RNN_H;
    unsigned* bar = (unsigned*)(h_pong + RNN_B * RNN_H);

    init_kernel<<<(RNN_B * RNN_H + 255) / 256, 256, 0, stream>>>(h0, h_ping, bar);

    dim3 g2((RNN_B * RNN_S) / BM, RNN_H / BN);   // 512 x 4
    gemm_pre_kernel<<<g2, 256, 0, stream>>>(x, w_ih, b_ih, b_hh, out);

    const size_t smem_bytes = (size_t)(64 * WSTR + 64 * HSTR) * sizeof(float);
    hipFuncSetAttribute(reinterpret_cast<const void*>(&rnn_steps_kernel),
                        hipFuncAttributeMaxDynamicSharedMemorySize,
                        (int)smem_bytes);
    rnn_steps_kernel<<<NWG, 256, smem_bytes, stream>>>(w_hh, out, h_ping, h_pong, bar);
}
